// SelfAttention_16750372454632
// MI455X (gfx1250) — matrix operand order repaired and checked by reading, not compiled, not run on hardware
//
#include <hip/hip_runtime.h>

// ---------------------------------------------------------------------------
// Self-attention (BERT-style) for MI455X / gfx1250.
// f16 inputs + f32 accumulate via V_WMMA_F32_16X16X32_F16.
// GEMMs: 4x4 fragment register blocking (1 b128 load per WMMA, L2-resident).
// Attention: 32 queries/wave flash loop, direct-exp softmax (scores are O(1)
// by construction; mask bias only drives them negative), deferred row-sum.
// ---------------------------------------------------------------------------

typedef __attribute__((ext_vector_type(16))) _Float16 v16h;
typedef __attribute__((ext_vector_type(8)))  _Float16 v8h;
typedef __attribute__((ext_vector_type(8)))  float    v8f;

constexpr int HID = 768;
constexpr int NH  = 12;
constexpr int HD  = 64;     // head dim
constexpr int BB  = 4;
constexpr int SS  = 2048;
constexpr int MM  = BB * SS;        // 8192 tokens
constexpr int W2  = HID * HID;      // 589824 weight elements

union AFrag { v16h v; v8h h[2]; };

// ---- fragment loaders (contiguous K order, same for A and B) ----------
// A 16x32 f16: lanes 0-15 = rows, VGPR0-3: K=kb..kb+7, VGPR4-7: K=kb+8..kb+15
// where kb = 16*(lane>>4).
__device__ __forceinline__ v16h wmma_ld_a(const _Float16* base, int ld,
                                          int row0, int k0, int lane) {
  int m  = lane & 15;
  int kb = (lane >> 4) * 16;
  const _Float16* p = base + (size_t)(row0 + m) * ld + k0 + kb;
  AFrag f;
  f.h[0] = *reinterpret_cast<const v8h*>(p);       // K = kb   .. kb+7
  f.h[1] = *reinterpret_cast<const v8h*>(p + 8);   // K = kb+8 .. kb+15
  return f.v;
}

// B 32x16 f16: lane = column n; lanes 0-15 hold K=0..15, lanes 16-31 K=16..31
// (contiguous K per lane). `base` rows are indexed by n, contiguous over K
// (i.e. B = rowMajor(base)^T, exactly the x @ W.T pattern).
__device__ __forceinline__ v16h wmma_ld_b_rowk(const _Float16* base, int ld,
                                               int n0, int k0, int lane) {
  int n  = lane & 15;
  int kb = (lane >> 4) * 16;
  const _Float16* p = base + (size_t)(n0 + n) * ld + k0 + kb;
  AFrag f;
  f.h[0] = *reinterpret_cast<const v8h*>(p);
  f.h[1] = *reinterpret_cast<const v8h*>(p + 8);
  return f.v;
}

__device__ __forceinline__ v8f wmma_f16(v16h a, v16h b, v8f c) {
  return __builtin_amdgcn_wmma_f32_16x16x32_f16(
      /*neg_a=*/false, a, /*neg_b=*/false, b,
      /*c_mod=*/(short)0, c, /*reuse_a=*/false, /*reuse_b=*/false);
}

__device__ __forceinline__ float red_sum16(float x) {
#pragma unroll
  for (int m = 8; m >= 1; m >>= 1) x += __shfl_xor(x, m, 32);
  return x;
}

// ---------------------------------------------------------------------------
// f32 -> f16 convert (grid-stride)
// ---------------------------------------------------------------------------
__global__ void cvt_f32_f16(const float* __restrict__ src,
                            _Float16* __restrict__ dst, int n) {
  for (int i = blockIdx.x * blockDim.x + threadIdx.x; i < n;
       i += gridDim.x * blockDim.x)
    dst[i] = (_Float16)src[i];
}

// ---------------------------------------------------------------------------
// QKV projection: C[8192 x 2304] = X * [Wq;Wk;Wv]^T + bias, scattered to
// Q[b,h,s,d], K[b,h,s,d], Vt[b,h,d,s] in f16.
// Grid: (18 n-tiles, 64 m-tiles), 128 threads = 4 waves in 2x2, each wave a
// 64x64 sub-tile = 4x4 fragments -> 16 WMMA per 16 b128 loads per k-step.
// ---------------------------------------------------------------------------
__global__ void __launch_bounds__(128)
qkv_gemm(const _Float16* __restrict__ X,
         const _Float16* __restrict__ Wq, const _Float16* __restrict__ Wk,
         const _Float16* __restrict__ Wv,
         const float* __restrict__ bq, const float* __restrict__ bk,
         const float* __restrict__ bv,
         _Float16* __restrict__ Qo, _Float16* __restrict__ Ko,
         _Float16* __restrict__ Vt) {
  const int lane = threadIdx.x & 31;
  const int wave = threadIdx.x >> 5;
  const int row0  = blockIdx.y * 128 + (wave & 1) * 64;   // token base
  const int ncol0 = blockIdx.x * 128 + (wave >> 1) * 64;  // col in [0,2304)

  // Q/K/V boundaries (768,1536) are multiples of 64 -> uniform per wave.
  const int which = ncol0 / HID;
  const int nloc0 = ncol0 - which * HID;
  const _Float16* Wm   = (which == 0) ? Wq : (which == 1) ? Wk : Wv;
  const float*    bias = (which == 0) ? bq : (which == 1) ? bk : bv;

  v8f acc[4][4] = {};
  for (int k0 = 0; k0 < HID; k0 += 32) {
    v16h a[4], bf[4];
#pragma unroll
    for (int i = 0; i < 4; ++i)
      a[i] = wmma_ld_a(X, HID, row0 + 16 * i, k0, lane);
#pragma unroll
    for (int j = 0; j < 4; ++j)
      bf[j] = wmma_ld_b_rowk(Wm, HID, nloc0 + 16 * j, k0, lane);
#pragma unroll
    for (int i = 0; i < 4; ++i)
#pragma unroll
      for (int j = 0; j < 4; ++j)
        acc[i][j] = wmma_f16(a[i], bf[j], acc[i][j]);
  }

  const int g = lane >> 4, nl = lane & 15;
#pragma unroll
  for (int j = 0; j < 4; ++j) {
    const int hd = nloc0 + j * 16 + nl;    // col within 768
    const float bv_ = bias[hd];
    const int h = hd >> 6, d = hd & 63;
#pragma unroll
    for (int i = 0; i < 4; ++i)
#pragma unroll
      for (int v = 0; v < 8; ++v) {
        const int tok = row0 + i * 16 + v + 8 * g;
        const int b = tok >> 11, s = tok & (SS - 1);
        const size_t bh = (size_t)(b * NH + h);
        const _Float16 hv = (_Float16)(acc[i][j][v] + bv_);
        if (which == 0)      Qo[(bh * SS + s) * HD + d] = hv;
        else if (which == 1) Ko[(bh * SS + s) * HD + d] = hv;
        else                 Vt[(bh * HD + d) * SS + s] = hv;  // transposed
      }
  }
}

// ---------------------------------------------------------------------------
// Flash attention: one wave per (b,h, 32-query tile). Direct-exp softmax in
// f32 (no running max needed: scores are O(1), mask only subtracts), row sums
// accumulated per-lane and reduced once at the end. Score + PV matmuls in
// WMMA f16; P bounces through LDS (D-layout -> A-layout).
// Grid: (S/32 = 64, B*NH = 48), blockDim = 32.
// ---------------------------------------------------------------------------
__global__ void __launch_bounds__(32)
attn_kernel(const _Float16* __restrict__ Q, const _Float16* __restrict__ K,
            const _Float16* __restrict__ Vt, const float* __restrict__ mask,
            _Float16* __restrict__ Ctx) {
  __shared__ __align__(16) _Float16 lds_p[32 * 32];  // 2 KB P tile

  const int lane = threadIdx.x & 31;
  const int q0 = blockIdx.x * 32;
  const int bh = blockIdx.y;
  const int b = bh / NH, h = bh - b * NH;
  const _Float16* Qb = Q  + (size_t)bh * SS * HD;
  const _Float16* Kb = K  + (size_t)bh * SS * HD;
  const _Float16* Vb = Vt + (size_t)bh * HD * SS;
  const float*    mr = mask + (size_t)b * SS;

  const int g = lane >> 4, nl = lane & 15;

  // Q fragments: 2 query rows x 2 d-chunks, loaded once.
  v16h aq[2][2];
#pragma unroll
  for (int qi = 0; qi < 2; ++qi)
#pragma unroll
    for (int c = 0; c < 2; ++c)
      aq[qi][c] = wmma_ld_a(Qb, HD, q0 + qi * 16, c * 32, lane);

  v8f acc[2][4] = {};         // 32q x 64d f32 accumulator
  float lacc[2][8] = {};      // per-lane partial exp-sums

  for (int kt = 0; kt < SS; kt += 32) {
    // K fragments shared by both query-row fragments.
    const v16h bk00 = wmma_ld_b_rowk(Kb, HD, kt,      0,  lane);
    const v16h bk01 = wmma_ld_b_rowk(Kb, HD, kt,      32, lane);
    const v16h bk10 = wmma_ld_b_rowk(Kb, HD, kt + 16, 0,  lane);
    const v16h bk11 = wmma_ld_b_rowk(Kb, HD, kt + 16, 32, lane);

    const float bias0 = (1.0f - mr[kt + nl])      * -10000.0f;
    const float bias1 = (1.0f - mr[kt + 16 + nl]) * -10000.0f;

#pragma unroll
    for (int qi = 0; qi < 2; ++qi) {
      v8f s0 = {}, s1 = {};
      s0 = wmma_f16(aq[qi][0], bk00, s0);
      s0 = wmma_f16(aq[qi][1], bk01, s0);
      s1 = wmma_f16(aq[qi][0], bk10, s1);
      s1 = wmma_f16(aq[qi][1], bk11, s1);
#pragma unroll
      for (int v = 0; v < 8; ++v) {
        const float p0 = __expf(s0[v] * 0.125f + bias0);
        const float p1 = __expf(s1[v] * 0.125f + bias1);
        lacc[qi][v] += p0 + p1;
        const int r = qi * 16 + v + 8 * g;
        lds_p[r * 32 + nl]      = (_Float16)p0;
        lds_p[r * 32 + 16 + nl] = (_Float16)p1;
      }
    }
    asm volatile("s_wait_dscnt 0" ::: "memory");   // single wave: order ds ops

    // reload P as A fragments; accumulate P*V per 16-wide d block.
    const v16h ap0 = wmma_ld_a(lds_p, 32, 0,  0, lane);
    const v16h ap1 = wmma_ld_a(lds_p, 32, 16, 0, lane);
#pragma unroll
    for (int db = 0; db < 4; ++db) {
      // B 32x16: column n -> d = db*16+n, K = key (contiguous in Vt rows)
      AFrag f;
      const _Float16* p = Vb + (size_t)(db * 16 + nl) * SS + kt + 16 * g;
      f.h[0] = *reinterpret_cast<const v8h*>(p);
      f.h[1] = *reinterpret_cast<const v8h*>(p + 8);
      acc[0][db] = wmma_f16(ap0, f.v, acc[0][db]);
      acc[1][db] = wmma_f16(ap1, f.v, acc[1][db]);
    }
  }

  // one cross-lane row-sum reduction at the end, normalize, store f16 ctx.
#pragma unroll
  for (int qi = 0; qi < 2; ++qi)
#pragma unroll
    for (int v = 0; v < 8; ++v) {
      const float inv = 1.0f / red_sum16(lacc[qi][v]);
      const int s = q0 + qi * 16 + v + 8 * g;
      _Float16* dst = Ctx + ((size_t)b * SS + s) * HID + h * HD;
#pragma unroll
      for (int db = 0; db < 4; ++db)
        dst[db * 16 + nl] = (_Float16)(acc[qi][db][v] * inv);
    }
}

// ---------------------------------------------------------------------------
// Output projection: out[8192 x 768] = Ctx * Wo^T + bo  (f32 output)
// Grid: (6, 64), 128 threads = 4 waves in 2x2, 128x128 tile, 4x4 fragments.
// ---------------------------------------------------------------------------
__global__ void __launch_bounds__(128)
out_gemm(const _Float16* __restrict__ Ctx, const _Float16* __restrict__ Wo,
         const float* __restrict__ bo, float* __restrict__ out) {
  const int lane = threadIdx.x & 31;
  const int wave = threadIdx.x >> 5;
  const int row0 = blockIdx.y * 128 + (wave & 1) * 64;
  const int col0 = blockIdx.x * 128 + (wave >> 1) * 64;

  v8f acc[4][4] = {};
  for (int k0 = 0; k0 < HID; k0 += 32) {
    v16h a[4], bf[4];
#pragma unroll
    for (int i = 0; i < 4; ++i)
      a[i] = wmma_ld_a(Ctx, HID, row0 + 16 * i, k0, lane);
#pragma unroll
    for (int j = 0; j < 4; ++j)
      bf[j] = wmma_ld_b_rowk(Wo, HID, col0 + 16 * j, k0, lane);
#pragma unroll
    for (int i = 0; i < 4; ++i)
#pragma unroll
      for (int j = 0; j < 4; ++j)
        acc[i][j] = wmma_f16(a[i], bf[j], acc[i][j]);
  }

  const int g = lane >> 4, nl = lane & 15;
#pragma unroll
  for (int j = 0; j < 4; ++j) {
    const int col = col0 + j * 16 + nl;
    const float bv_ = bo[col];
#pragma unroll
    for (int i = 0; i < 4; ++i)
#pragma unroll
      for (int v = 0; v < 8; ++v) {
        const int row = row0 + i * 16 + v + 8 * g;
        out[(size_t)row * HID + col] = acc[i][j][v] + bv_;
      }
  }
}

// ---------------------------------------------------------------------------
extern "C" void kernel_launch(void* const* d_in, const int* in_sizes, int n_in,
                              void* d_out, int out_size, void* d_ws,
                              size_t ws_size, hipStream_t stream) {
  const float* hs   = (const float*)d_in[0];
  const float* amsk = (const float*)d_in[1];
  const float* Wq   = (const float*)d_in[2];
  const float* bq   = (const float*)d_in[3];
  const float* Wk   = (const float*)d_in[4];
  const float* bk   = (const float*)d_in[5];
  const float* Wv   = (const float*)d_in[6];
  const float* bv   = (const float*)d_in[7];
  const float* Wo   = (const float*)d_in[8];
  const float* bo   = (const float*)d_in[9];
  float* out = (float*)d_out;

  // workspace partition (f16 halves)
  _Float16* w = (_Float16*)d_ws;
  _Float16* Xh  = w; w += (size_t)MM * HID;
  _Float16* Wqh = w; w += W2;
  _Float16* Wkh = w; w += W2;
  _Float16* Wvh = w; w += W2;
  _Float16* Woh = w; w += W2;
  _Float16* Qh  = w; w += (size_t)MM * HID;
  _Float16* Kh  = w; w += (size_t)MM * HID;
  _Float16* Vth = w; w += (size_t)MM * HID;
  _Float16* Ctx = w; w += (size_t)MM * HID;

  const int nX = MM * HID;
  cvt_f32_f16<<<(nX + 255) / 256, 256, 0, stream>>>(hs, Xh, nX);
  cvt_f32_f16<<<(W2 + 255) / 256, 256, 0, stream>>>(Wq, Wqh, W2);
  cvt_f32_f16<<<(W2 + 255) / 256, 256, 0, stream>>>(Wk, Wkh, W2);
  cvt_f32_f16<<<(W2 + 255) / 256, 256, 0, stream>>>(Wv, Wvh, W2);
  cvt_f32_f16<<<(W2 + 255) / 256, 256, 0, stream>>>(Wo, Woh, W2);

  qkv_gemm<<<dim3(3 * HID / 128, MM / 128), 128, 0, stream>>>(
      Xh, Wqh, Wkh, Wvh, bq, bk, bv, Qh, Kh, Vth);

  attn_kernel<<<dim3(SS / 32, BB * NH), 32, 0, stream>>>(Qh, Kh, Vth, amsk,
                                                         Ctx);

  out_gemm<<<dim3(HID / 128, MM / 128), 128, 0, stream>>>(Ctx, Woh, bo, out);
}
